// SpikingAudioNet_24060406792396
// MI455X (gfx1250) — matrix operand order repaired and checked by reading, not compiled, not run on hardware
//
#include <hip/hip_runtime.h>
#include <cstdint>

// ============================================================================
// SNN (Leaky LIF, 3 layers) for MI455X / gfx1250, wave32 + WMMA bf16.
//
// Kernel W: one-time W1 f32 -> bf16 into scratch (removes all per-k-step
//           conversion VALU from the GEMM hot loop).
// Kernel A: cur1_all = x @ W1^T + b1. M=32/N=512 tile per block (8 waves),
//           each wave: 2 M-tiles x 4 N-tiles -> 8 v8f accumulators; every
//           B fragment feeds 2 WMMAs. Bandwidth-bound: ~210 MB HBM ~ 9 us.
// Kernel B: 100-step recurrent scan, 16 blocks each owning 16 batch rows;
//           W2 bf16 resident in LDS (256 KB < 320 KB/WGP), layer2 via WMMA
//           with m2 state kept directly in the WMMA C/D register layout.
// ============================================================================

typedef __attribute__((ext_vector_type(16))) __bf16 v16bf;
typedef __attribute__((ext_vector_type(8)))  float  v8f;

struct U4 { unsigned int x, y, z, w; };          // trivial 16B chunk (union-safe)
union FragU { U4 u[2]; v16bf v; };               // 32B <-> 16 bf16 lanes

__device__ __forceinline__ unsigned int bf16of(float f) {
  unsigned int u = __float_as_uint(f);
  return ((u + 0x7FFFu + ((u >> 16) & 1u)) >> 16) & 0xFFFFu;   // RNE
}
__device__ __forceinline__ unsigned int pk2bf(float lo, float hi) {
  return bf16of(lo) | (bf16of(hi) << 16);
}
__device__ __forceinline__ v8f wmma_bf16(v16bf a, v16bf b, v8f c) {
  // (neg_a, A, neg_b, B, c_mod, C, reuse_a, reuse_b)
  return __builtin_amdgcn_wmma_f32_16x16x32_bf16(false, a, false, b, (short)0, c,
                                                 false, false);
}

#define BETA_F   0.9f
#define THRESH_F 1.0f
#define T_STEPS  100
#define B_SZ     256
#define D_IN     1024
#define H1_SZ    512
#define H2_SZ    256
#define C_SZ     10

// ---------------------------------------------------------------------------
// Kernel W: W1 [512][1024] f32 -> bf16 (one-time, 0.5M elems). 8 elems/thread.
// ---------------------------------------------------------------------------
__global__ __launch_bounds__(256) void cvt_w1_bf16(
    const float* __restrict__ W1, unsigned int* __restrict__ W1bf)
{
  const int i = blockIdx.x * 256 + threadIdx.x;     // 65536 threads
  const float4* s = (const float4*)W1 + (size_t)i * 2;
  float4 a = s[0], b = s[1];
  U4 o{pk2bf(a.x, a.y), pk2bf(a.z, a.w), pk2bf(b.x, b.y), pk2bf(b.z, b.w)};
  ((U4*)W1bf)[i] = o;
}

// ---------------------------------------------------------------------------
// Kernel A: cur1[r, n] = sum_k x[r,k]*W1[n,k] + b1[n],  r in [0,25600)
// Block: 256 thr (8 waves), tile M=32 x N=512. Wave w: N-tiles 4w..4w+3,
// both M-tiles. x tile staged as bf16 in dynamic LDS (64 KB); W1 fragments
// are raw bf16 loads from the pre-converted scratch (L2-hot, 1 MB total).
// ---------------------------------------------------------------------------
__global__ __launch_bounds__(256) void snn_fc1_gemm(
    const float* __restrict__ x,             // [25600][1024] f32
    const unsigned int* __restrict__ W1bf,   // [512][1024] bf16 (as uint pairs)
    const float* __restrict__ b1,            // [512]
    float* __restrict__ cur1)                // [25600][512] f32
{
  extern __shared__ char smem[];             // 32 x 1024 bf16 = 64 KB
  unsigned int* s_x = (unsigned int*)smem;
  const int tid   = threadIdx.x;
  const int mrow0 = blockIdx.x * 32;

  // ---- stage + convert x tile: thread -> 128 contiguous cols of one row ----
  {
    const int r  = tid >> 3;             // 0..31
    const int c0 = (tid & 7) * 128;      // 0..896
    const float4* src = (const float4*)(x + (size_t)(mrow0 + r) * D_IN + c0);
    unsigned int* dst = s_x + (r * D_IN + c0) / 2;
#pragma unroll 8
    for (int i = 0; i < 32; ++i) {
      float4 v = src[i];
      dst[2 * i + 0] = pk2bf(v.x, v.y);
      dst[2 * i + 1] = pk2bf(v.z, v.w);
    }
  }
  __syncthreads();

  const int lane  = tid & 31;
  const int wave  = tid >> 5;
  const int lm    = lane & 15;
  const int khA   = (lane < 16) ? 0 : 8;    // A frag K base (ISA 16-bit A layout)
  const int khB   = (lane < 16) ? 0 : 8;    // B frag K base (ISA 16-bit B layout)
  const int mbase = (lane < 16) ? 0 : 8;    // C/D rows this lane owns

  v8f acc[2][4];
#pragma unroll
  for (int m = 0; m < 2; ++m)
#pragma unroll
    for (int j = 0; j < 4; ++j)
      acc[m][j] = v8f{0.f,0.f,0.f,0.f,0.f,0.f,0.f,0.f};

  for (int ko = 0; ko < D_IN; ko += 32) {
    // Two A fragments (M-tile 0: rows 0..15, M-tile 1: rows 16..31)
    FragU a0, a1;
    const char* ar0 = (const char*)s_x + (size_t)(lm * D_IN + ko + khA) * 2;
    const char* ar1 = ar0 + (size_t)16 * D_IN * 2;
    a0.u[0] = *(const U4*)(ar0);  a0.u[1] = *(const U4*)(ar0 + 32);
    a1.u[0] = *(const U4*)(ar1);  a1.u[1] = *(const U4*)(ar1 + 32);
#pragma unroll
    for (int j = 0; j < 4; ++j) {
      const int n = (wave * 4 + j) * 16 + lm;          // output column
      const char* wr = (const char*)W1bf + (size_t)(n * D_IN + ko + khB) * 2;
      FragU bfr;
      bfr.u[0] = *(const U4*)(wr);
      bfr.u[1] = *(const U4*)(wr + 32);
      acc[0][j] = wmma_bf16(a0.v, bfr.v, acc[0][j]);   // B frag reused twice
      acc[1][j] = wmma_bf16(a1.v, bfr.v, acc[1][j]);
    }
  }

  // ---- epilogue: + bias, store (lane owns column n, rows mbase..mbase+7) ----
#pragma unroll
  for (int j = 0; j < 4; ++j) {
    const int n = (wave * 4 + j) * 16 + lm;
    const float bias = b1[n];
#pragma unroll
    for (int m = 0; m < 2; ++m)
#pragma unroll
      for (int r = 0; r < 8; ++r)
        cur1[(size_t)(mrow0 + m * 16 + mbase + r) * H1_SZ + n] = acc[m][j][r] + bias;
  }
}

// ---------------------------------------------------------------------------
// Kernel B: recurrent scan. 16 blocks x 256 thr; block g owns batch rows
// [16g,16g+16). LDS: W2 bf16 (256K) | s1 bf16 (16K) | s2 f32 (16K) |
// W3 f32 (10K) | b2 (1K) | b3. m1: 32 f32/thread regs; m2: two v8f in WMMA
// layout per lane; m3: 1 f32 in threads 0..159.
// ---------------------------------------------------------------------------
__global__ __launch_bounds__(256) void snn_scan(
    const float* __restrict__ cur1,  // [100*256][512]
    const float* __restrict__ W2,    // [256][512]
    const float* __restrict__ b2g,   // [256]
    const float* __restrict__ W3,    // [10][256]
    const float* __restrict__ b3g,   // [10]
    float* __restrict__ out)         // [256][10]
{
  extern __shared__ char smem[];
  unsigned int* sW2 = (unsigned int*)(smem);                      // 262144 B
  unsigned int* sS1 = (unsigned int*)(smem + 262144);             //  16384 B
  float*        sS2 = (float*)(smem + 262144 + 16384);            //  16384 B
  float*        sW3 = (float*)(smem + 262144 + 32768);            //  10240 B
  float*        sB2 = (float*)(smem + 262144 + 32768 + 10240);    //   1024 B
  float*        sB3 = (float*)(smem + 262144 + 32768 + 10240 + 1024);

  const int tid = threadIdx.x;
  const int g   = blockIdx.x;                 // batch tile index

  // ---- one-time LDS fills ----
  {   // W2 f32 -> bf16, one row (512 elems) per thread (one-time cost)
    const float4* src = (const float4*)(W2 + (size_t)tid * H1_SZ);
    unsigned int* dst = sW2 + tid * (H1_SZ / 2);
#pragma unroll 4
    for (int i = 0; i < 128; ++i) {
      float4 v = src[i];
      dst[2 * i + 0] = pk2bf(v.x, v.y);
      dst[2 * i + 1] = pk2bf(v.z, v.w);
    }
  }
  for (int i = tid; i < C_SZ * H2_SZ; i += 256) sW3[i] = W3[i];
  sB2[tid] = b2g[tid];
  if (tid < C_SZ) sB3[tid] = b3g[tid];
  __syncthreads();

  // ---- state ----
  const int bloc = tid >> 4;                  // local batch row for m1 (0..15)
  const int h1c  = (tid & 15) * 32;           // 32-wide h1 slice for m1
  float m1[32];
#pragma unroll
  for (int i = 0; i < 32; ++i) m1[i] = 0.f;

  const int lane  = tid & 31;
  const int wave  = tid >> 5;
  const int lm    = lane & 15;
  const int mbase = (lane < 16) ? 0 : 8;
  const int khA   = (lane < 16) ? 0 : 8;
  const int khB   = (lane < 16) ? 0 : 8;
  const int n0    = (wave * 2 + 0) * 16 + lm; // this lane's two h2 columns
  const int n1    = (wave * 2 + 1) * 16 + lm;
  v8f m2a = v8f{0.f,0.f,0.f,0.f,0.f,0.f,0.f,0.f};
  v8f m2b = v8f{0.f,0.f,0.f,0.f,0.f,0.f,0.f,0.f};

  float m3 = 0.f;
  const int b3row = tid / C_SZ, c3 = tid % C_SZ;   // valid for tid < 160

  const float bias0 = sB2[n0], bias1 = sB2[n1];

  for (int t = 0; t < T_STEPS; ++t) {
    // -------- layer 1: m1 LIF update, emit s1 --------
    // spikes are exactly 0.0/1.0 -> bf16 encode is a select of 0x3F80
    const float* cr = cur1 + ((size_t)(t * B_SZ + g * 16 + bloc)) * H1_SZ + h1c;
    if (t + 1 < T_STEPS)        // gfx1250 global_prefetch for next timestep
      __builtin_prefetch(cur1 + ((size_t)((t + 1) * B_SZ + g * 16 + bloc)) * H1_SZ + h1c, 0, 3);
    float cl[32];
    const float4* cr4 = (const float4*)cr;
#pragma unroll
    for (int i = 0; i < 8; ++i) {
      float4 v = cr4[i];
      cl[4*i] = v.x; cl[4*i+1] = v.y; cl[4*i+2] = v.z; cl[4*i+3] = v.w;
    }
    unsigned int sp[16];
#pragma unroll
    for (int i = 0; i < 32; i += 2) {
      float r0 = (m1[i]   > THRESH_F) ? THRESH_F : 0.f;   // reset from incoming mem
      float r1 = (m1[i+1] > THRESH_F) ? THRESH_F : 0.f;
      m1[i]   = BETA_F * m1[i]   + cl[i]   - r0;
      m1[i+1] = BETA_F * m1[i+1] + cl[i+1] - r1;
      unsigned int lo = (m1[i]   - THRESH_F > 0.f) ? 0x00003F80u : 0u;  // bf16(1.0)
      unsigned int hi = (m1[i+1] - THRESH_F > 0.f) ? 0x3F800000u : 0u;  // <<16
      sp[i / 2] = lo | hi;
    }
    {
      U4* d = (U4*)((char*)sS1 + (size_t)(bloc * H1_SZ + h1c) * 2);
      d[0] = U4{sp[0],  sp[1],  sp[2],  sp[3]};
      d[1] = U4{sp[4],  sp[5],  sp[6],  sp[7]};
      d[2] = U4{sp[8],  sp[9],  sp[10], sp[11]};
      d[3] = U4{sp[12], sp[13], sp[14], sp[15]};
    }
    __syncthreads();

    // -------- layer 2: [16 x 512] @ W2^T via WMMA; m2 LIF in C/D layout ----
    v8f acc0 = v8f{0.f,0.f,0.f,0.f,0.f,0.f,0.f,0.f};
    v8f acc1 = v8f{0.f,0.f,0.f,0.f,0.f,0.f,0.f,0.f};
#pragma unroll
    for (int ko = 0; ko < H1_SZ; ko += 32) {
      FragU af;
      const char* ar = (const char*)sS1 + (size_t)(lm * H1_SZ + ko + khA) * 2;
      af.u[0] = *(const U4*)(ar);
      af.u[1] = *(const U4*)(ar + 32);
      FragU bf0, bf1;
      const char* br0 = (const char*)sW2 + (size_t)(n0 * H1_SZ + ko + khB) * 2;
      const char* br1 = (const char*)sW2 + (size_t)(n1 * H1_SZ + ko + khB) * 2;
      bf0.u[0] = *(const U4*)(br0);  bf0.u[1] = *(const U4*)(br0 + 32);
      bf1.u[0] = *(const U4*)(br1);  bf1.u[1] = *(const U4*)(br1 + 32);
      acc0 = wmma_bf16(af.v, bf0.v, acc0);
      acc1 = wmma_bf16(af.v, bf1.v, acc1);
    }
#pragma unroll
    for (int r = 0; r < 8; ++r) {
      float c20 = acc0[r] + bias0;
      float rs0 = (m2a[r] > THRESH_F) ? THRESH_F : 0.f;
      float mn0 = BETA_F * m2a[r] + c20 - rs0;
      m2a[r] = mn0;
      sS2[(mbase + r) * H2_SZ + n0] = (mn0 - THRESH_F > 0.f) ? 1.f : 0.f;

      float c21 = acc1[r] + bias1;
      float rs1 = (m2b[r] > THRESH_F) ? THRESH_F : 0.f;
      float mn1 = BETA_F * m2b[r] + c21 - rs1;
      m2b[r] = mn1;
      sS2[(mbase + r) * H2_SZ + n1] = (mn1 - THRESH_F > 0.f) ? 1.f : 0.f;
    }
    __syncthreads();

    // -------- layer 3: 16x10 <= [16 x 256] @ W3^T, VALU (tiny), m3 LIF -----
    if (tid < 16 * C_SZ) {
      const float4* s4 = (const float4*)(sS2 + b3row * H2_SZ);
      const float4* w4 = (const float4*)(sW3 + c3 * H2_SZ);
      float a = sB3[c3];
#pragma unroll 8
      for (int k = 0; k < H2_SZ / 4; ++k) {
        float4 s = s4[k], w = w4[k];
        a = fmaf(s.x, w.x, a); a = fmaf(s.y, w.y, a);
        a = fmaf(s.z, w.z, a); a = fmaf(s.w, w.w, a);
      }
      float rs = (m3 > THRESH_F) ? THRESH_F : 0.f;
      m3 = BETA_F * m3 + a - rs;
    }
    __syncthreads();   // protect sS1/sS2 reuse next iteration
  }

  if (tid < 16 * C_SZ)
    out[(size_t)(g * 16 + b3row) * C_SZ + c3] = m3;
}

// ---------------------------------------------------------------------------
extern "C" void kernel_launch(void* const* d_in, const int* in_sizes, int n_in,
                              void* d_out, int out_size, void* d_ws, size_t ws_size,
                              hipStream_t stream) {
  (void)in_sizes; (void)n_in; (void)out_size;
  const float* x  = (const float*)d_in[0];
  const float* W1 = (const float*)d_in[1];
  const float* b1 = (const float*)d_in[2];
  const float* W2 = (const float*)d_in[3];
  const float* b2 = (const float*)d_in[4];
  const float* W3 = (const float*)d_in[5];
  const float* b3 = (const float*)d_in[6];
  float* out = (float*)d_out;

  // Scratch layout: cur1 f32 [25600*512] (52,428,800 B) | W1 bf16 (1,048,576 B)
  const size_t cur1_bytes = (size_t)T_STEPS * B_SZ * H1_SZ * sizeof(float);
  const size_t w1bf_bytes = (size_t)H1_SZ * D_IN * 2;
  if (ws_size < cur1_bytes + w1bf_bytes) return;   // deterministic no-op guard
  float*        cur1 = (float*)d_ws;
  unsigned int* W1bf = (unsigned int*)((char*)d_ws + cur1_bytes);

  // W: 512*1024/8 = 65536 threads
  cvt_w1_bf16<<<256, 256, 0, stream>>>(W1, W1bf);

  // A: 25600/32 = 800 M-tiles, 64 KB dynamic LDS for the x tile
  const size_t shA = (size_t)32 * D_IN * 2;
  hipFuncSetAttribute(reinterpret_cast<const void*>(snn_fc1_gemm),
                      hipFuncAttributeMaxDynamicSharedMemorySize, (int)shA);
  snn_fc1_gemm<<<800, 256, shA, stream>>>(x, W1bf, b1, cur1);

  // B: dynamic LDS = W2(bf16) + s1 + s2 + W3 + b2 + b3pad
  const size_t shB = 262144 + 16384 + 16384 + 10240 + 1024 + 64;   // 306240
  hipFuncSetAttribute(reinterpret_cast<const void*>(snn_scan),
                      hipFuncAttributeMaxDynamicSharedMemorySize, (int)shB);
  snn_scan<<<16, 256, shB, stream>>>(cur1, W2, b2, W3, b3, out);
}
